// MultiHeadAttentionWithOrganic_67010079752830
// MI455X (gfx1250) — matrix operand order repaired and checked by reading, not compiled, not run on hardware
//
#include <hip/hip_runtime.h>

// ---------------------------------------------------------------------------
// MultiHeadAttentionWithOrganic for MI455X (gfx1250, CDNA5, wave32, WMMA)
// B=4, S=2048, D=512, H=8, DK=64
// ---------------------------------------------------------------------------

typedef __attribute__((ext_vector_type(16))) __bf16 v16bf;
typedef __attribute__((ext_vector_type(8)))  float  v8f;

#define B_  4
#define S_  2048
#define D_  512
#define H_  8
#define DK_ 64

// ---------------- WMMA fragment loaders (contiguous k order, same for A and B) ----------------
// A-matrix 16x32 bf16: lane l -> row m=l&15; group g=l>>4;
//   elements 0..7  = K[kbase + 16g     .. +8)
//   elements 8..15 = K[kbase + 16g + 8 .. +8)
static __device__ __forceinline__ v16bf frag_a_f32(const float* base, int stride,
                                                   int row0, int kbase, int lane) {
  int m = lane & 15, g = lane >> 4;
  const float* p0 = base + (size_t)(row0 + m) * stride + kbase + 16 * g;
  const float* p1 = p0 + 8;
  v16bf a;
#pragma unroll
  for (int i = 0; i < 8; ++i) a[i] = (__bf16)p0[i];
#pragma unroll
  for (int i = 0; i < 8; ++i) a[8 + i] = (__bf16)p1[i];
  return a;
}

static __device__ __forceinline__ v16bf frag_a_bf16(const __bf16* base, int stride,
                                                    int row0, int kbase, int lane) {
  int m = lane & 15, g = lane >> 4;
  const __bf16* p0 = base + (size_t)(row0 + m) * stride + kbase + 16 * g;
  const __bf16* p1 = p0 + 8;
  v16bf a;
#pragma unroll
  for (int i = 0; i < 8; ++i) a[i] = p0[i];
#pragma unroll
  for (int i = 0; i < 8; ++i) a[8 + i] = p1[i];
  return a;
}

// B-matrix 32x16 bf16: lane l -> column n=l&15; group g=l>>4;
//   elements 0..15 = K[k0 + 16g .. +16)   (contiguous)
// Bmat[k][n] = base[(n0+n)*stride + k0 + k]
static __device__ __forceinline__ v16bf frag_b_f32(const float* base, int stride,
                                                   int n0, int k0, int lane) {
  int n = lane & 15, g = lane >> 4;
  const float* p = base + (size_t)(n0 + n) * stride + k0 + 16 * g;
  v16bf v;
#pragma unroll
  for (int i = 0; i < 16; ++i) v[i] = (__bf16)p[i];
  return v;
}

static __device__ __forceinline__ v16bf frag_b_bf16(const __bf16* base, int stride,
                                                    int n0, int k0, int lane) {
  int n = lane & 15, g = lane >> 4;
  const __bf16* p = base + (size_t)(n0 + n) * stride + k0 + 16 * g;
  v16bf v;
#pragma unroll
  for (int i = 0; i < 16; ++i) v[i] = p[i];
  return v;
}

// order-preserving float<->int for LDS integer atomic max
static __device__ __forceinline__ int f2ord(float f) {
  int i = __float_as_int(f);
  return (i >= 0) ? i : (i ^ 0x7fffffff);
}
static __device__ __forceinline__ float ord2f(int i) {
  return __int_as_float((i >= 0) ? i : (i ^ 0x7fffffff));
}

// ---------------------------------------------------------------------------
// Projection GEMM: C[m,n] = sum_k A[m,k] * W[n,k] + bias[n]   (M=8192,N=K=512)
// MODE 0: out bf16 [B,H,S,DK]   (Q, K)
// MODE 1: out bf16 [B,H,DK,S]   (V transposed, so P@V B-frags are contiguous)
// MODE 2: out f32  [M, N]       (final output projection)
// Block 256 = 8 waves, each wave owns one 16x16 tile; block tile 64(M)x32(N).
// ---------------------------------------------------------------------------
template <int MODE>
__global__ __launch_bounds__(256) void proj_gemm(const float* __restrict__ A,
                                                 const float* __restrict__ W,
                                                 const float* __restrict__ bias,
                                                 void* __restrict__ out) {
  int lane = threadIdx.x & 31;
  int w    = threadIdx.x >> 5;
  int m0 = blockIdx.y * 64 + (w >> 1) * 16;
  int n0 = blockIdx.x * 32 + (w & 1) * 16;

  v8f c = {};
#pragma unroll 4
  for (int k = 0; k < D_; k += 32) {
    v16bf av = frag_a_f32(A, D_, m0, k, lane);
    v16bf bv = frag_b_f32(W, D_, n0, k, lane);
    c = __builtin_amdgcn_wmma_f32_16x16x32_bf16(false, av, false, bv,
                                                (short)0, c, false, false);
  }

  int n_in = lane & 15, g = lane >> 4;
  int n = n0 + n_in;
  float bias_n = bias[n];
#pragma unroll
  for (int r = 0; r < 8; ++r) {
    int   m = m0 + r + 8 * g;          // C/D layout: VGPR r -> row r + 8*group
    float v = c[r] + bias_n;
    int bb = m >> 11, s = m & (S_ - 1);
    int hh = n >> 6,  dk = n & (DK_ - 1);
    if (MODE == 0) {
      ((__bf16*)out)[(((size_t)(bb * H_ + hh) * S_) + s) * DK_ + dk] = (__bf16)v;
    } else if (MODE == 1) {
      ((__bf16*)out)[(((size_t)(bb * H_ + hh) * DK_) + dk) * S_ + s] = (__bf16)v;
    } else {
      ((float*)out)[(size_t)m * D_ + n] = v;
    }
  }
}

// ---------------------------------------------------------------------------
// Fused attention: one workgroup per (b,h, 16-query-row tile).
// Pass1: scores -> row max.  Pass2: recompute -> row sum(exp).
// Pass3: recompute -> P, nontemporal-store attention (write-once),
//        LDS-bounce P to A-layout, WMMA-accumulate context = P @ V.
// ---------------------------------------------------------------------------
__global__ __launch_bounds__(256) void attn_fused(const __bf16* __restrict__ Qbf,
                                                  const __bf16* __restrict__ Kbf,
                                                  const __bf16* __restrict__ Vt,
                                                  const int*    __restrict__ mask,
                                                  const float*  __restrict__ isorg,
                                                  const float*  __restrict__ Worg,
                                                  const float*  __restrict__ borg,
                                                  float* __restrict__ attn,
                                                  float* __restrict__ ctx) {
  __shared__ int    rowmaxi[16];
  __shared__ float  rowsum[16];
  __shared__ float  rowbias[16];
  __shared__ __bf16 pbuf[8][16 * 32];     // per-wave P bounce tile (8 KB)
  __shared__ float  cpart[8][16][64];     // per-wave partial context (32 KB)

  int tid  = threadIdx.x;
  int lane = tid & 31;
  int w    = tid >> 5;
  int n_in = lane & 15, g = lane >> 4;

  int qt = blockIdx.x & (S_ / 16 - 1);    // 128 query tiles
  int bh = blockIdx.x >> 7;
  int b  = bh >> 3, h = bh & (H_ - 1);

  const __bf16* Qp = Qbf + (size_t)bh * S_ * DK_;
  const __bf16* Kp = Kbf + (size_t)bh * S_ * DK_;
  const __bf16* Vp = Vt  + (size_t)bh * DK_ * S_;
  const int*    mp = mask + (size_t)b * S_;
  float* attp = attn + ((size_t)bh * S_ + (size_t)qt * 16) * S_;

  if (tid < 16) {
    rowmaxi[tid] = f2ord(-3.0e38f);
    rowsum[tid]  = 0.0f;
    int s = qt * 16 + tid;
    rowbias[tid] = isorg[(size_t)b * S_ + s] * Worg[h] + borg[h];
  }
  __syncthreads();

  float rb[8];
#pragma unroll
  for (int r = 0; r < 8; ++r) rb[r] = rowbias[r + 8 * g];

  // Q fragments for this query tile (DK=64 -> two K=32 steps), reused all passes
  v16bf aq0 = frag_a_bf16(Qp, DK_, qt * 16, 0,  lane);
  v16bf aq1 = frag_a_bf16(Qp, DK_, qt * 16, 32, lane);

  // score tile for key-tile kt: s[r] for rows m=r+8g, column kt*16+n_in
  auto score_tile = [&](int kt, float* s8) {
    v16bf bk0 = frag_b_bf16(Kp, DK_, kt * 16, 0,  lane);
    v16bf bk1 = frag_b_bf16(Kp, DK_, kt * 16, 32, lane);
    v8f c = {};
    c = __builtin_amdgcn_wmma_f32_16x16x32_bf16(false, aq0, false, bk0, (short)0, c, false, false);
    c = __builtin_amdgcn_wmma_f32_16x16x32_bf16(false, aq1, false, bk1, (short)0, c, false, false);
    bool masked = (mp[kt * 16 + n_in] == 0);
#pragma unroll
    for (int r = 0; r < 8; ++r) {
      float s = c[r] * 0.125f + rb[r];            // /sqrt(64) + organic bias
      s8[r] = masked ? -1.0e9f : s;
    }
  };

  // ---- Pass 1: row max ----------------------------------------------------
  float rmax[8];
#pragma unroll
  for (int r = 0; r < 8; ++r) rmax[r] = -3.0e38f;
  for (int kt = w; kt < S_ / 16; kt += 8) {
    float s8[8];
    score_tile(kt, s8);
#pragma unroll
    for (int r = 0; r < 8; ++r) rmax[r] = fmaxf(rmax[r], s8[r]);
  }
#pragma unroll
  for (int r = 0; r < 8; ++r) atomicMax(&rowmaxi[r + 8 * g], f2ord(rmax[r]));
  __syncthreads();

  float rm[8];
#pragma unroll
  for (int r = 0; r < 8; ++r) rm[r] = ord2f(rowmaxi[r + 8 * g]);

  // ---- Pass 2: row sum of exp --------------------------------------------
  float rsum[8] = {0, 0, 0, 0, 0, 0, 0, 0};
  for (int kt = w; kt < S_ / 16; kt += 8) {
    float s8[8];
    score_tile(kt, s8);
#pragma unroll
    for (int r = 0; r < 8; ++r) rsum[r] += __expf(s8[r] - rm[r]);
  }
#pragma unroll
  for (int r = 0; r < 8; ++r) atomicAdd(&rowsum[r + 8 * g], rsum[r]);
  __syncthreads();

  float rinv[8];
#pragma unroll
  for (int r = 0; r < 8; ++r) rinv[r] = 1.0f / rowsum[r + 8 * g];

  // ---- Pass 3: P, streamed attention output, context = P @ V -------------
  v8f cacc[4];
#pragma unroll
  for (int nt = 0; nt < 4; ++nt) cacc[nt] = (v8f){};

  __bf16* pw = &pbuf[w][0];
  for (int ktp = w; ktp < S_ / 32; ktp += 8) {     // 32 keys per iteration
#pragma unroll
    for (int half = 0; half < 2; ++half) {
      int kt = ktp * 2 + half;
      float s8[8];
      score_tile(kt, s8);
#pragma unroll
      for (int r = 0; r < 8; ++r) {
        float p = __expf(s8[r] - rm[r]) * rinv[r];
        int   m = r + 8 * g;
        // write-once streaming store of the attention matrix (537 MB total)
        __builtin_nontemporal_store(p, &attp[(size_t)m * S_ + kt * 16 + n_in]);
        pw[m * 32 + half * 16 + n_in] = (__bf16)p;   // D-layout -> LDS
      }
    }
    // make this wave's LDS P-tile visible before re-loading in A-layout
    asm volatile("s_wait_dscnt 0" ::: "memory");
    __builtin_amdgcn_wave_barrier();

    v16bf pa = frag_a_bf16(pw, 32, 0, 0, lane);      // P tile 16x32 as A
#pragma unroll
    for (int nt = 0; nt < 4; ++nt) {
      // Bmat[k][n] = V[key = ktp*32+k][dk = nt*16+n] = Vt[(nt*16+n)*S + ktp*32+k]
      v16bf vb = frag_b_bf16(Vp, S_, nt * 16, ktp * 32, lane);
      cacc[nt] = __builtin_amdgcn_wmma_f32_16x16x32_bf16(false, pa, false, vb,
                                                         (short)0, cacc[nt], false, false);
    }
  }

  // combine 8 per-wave partial contexts
#pragma unroll
  for (int nt = 0; nt < 4; ++nt)
#pragma unroll
    for (int r = 0; r < 8; ++r) cpart[w][r + 8 * g][nt * 16 + n_in] = cacc[nt][r];
  __syncthreads();

  for (int i = tid; i < 16 * 64; i += 256) {
    int m = i >> 6, dk = i & 63;
    float sum = 0.0f;
#pragma unroll
    for (int ww = 0; ww < 8; ++ww) sum += cpart[ww][m][dk];
    ctx[((size_t)b * S_ + qt * 16 + m) * D_ + h * DK_ + dk] = sum;
  }
}

// ---------------------------------------------------------------------------
extern "C" void kernel_launch(void* const* d_in, const int* in_sizes, int n_in,
                              void* d_out, int out_size, void* d_ws, size_t ws_size,
                              hipStream_t stream) {
  (void)in_sizes; (void)n_in; (void)out_size; (void)ws_size;

  const float* query = (const float*)d_in[0];
  const float* key   = (const float*)d_in[1];
  const float* value = (const float*)d_in[2];
  const int*   mask  = (const int*)  d_in[3];
  const float* isorg = (const float*)d_in[4];
  const float* Wq = (const float*)d_in[5];   const float* bq = (const float*)d_in[6];
  const float* Wk = (const float*)d_in[7];   const float* bk = (const float*)d_in[8];
  const float* Wv = (const float*)d_in[9];   const float* bv = (const float*)d_in[10];
  const float* Wo = (const float*)d_in[11];  const float* bo = (const float*)d_in[12];
  const float* Worg = (const float*)d_in[13];
  const float* borg = (const float*)d_in[14];

  // workspace: Qbf | Kbf | Vt (bf16, 8 MB each) | ctx (f32, 16 MB)
  const size_t NE = (size_t)B_ * H_ * S_ * DK_;   // 4,194,304 elements
  char* wsb = (char*)d_ws;
  __bf16* Qbf = (__bf16*)(wsb);
  __bf16* Kbf = (__bf16*)(wsb + 2 * NE);
  __bf16* Vt  = (__bf16*)(wsb + 4 * NE);
  float*  ctx = (float*) (wsb + 6 * NE);

  float* outp = (float*)d_out;
  float* attn = outp + (size_t)B_ * S_ * D_;      // second tuple element

  dim3 pg(D_ / 32, (B_ * S_) / 64);               // 16 x 128 workgroups
  dim3 pb(256);

  proj_gemm<0><<<pg, pb, 0, stream>>>(query, Wq, bq, (void*)Qbf);
  proj_gemm<0><<<pg, pb, 0, stream>>>(key,   Wk, bk, (void*)Kbf);
  proj_gemm<1><<<pg, pb, 0, stream>>>(value, Wv, bv, (void*)Vt);

  attn_fused<<<B_ * H_ * (S_ / 16), 256, 0, stream>>>(Qbf, Kbf, Vt, mask, isorg,
                                                      Worg, borg, attn, ctx);

  proj_gemm<2><<<pg, pb, 0, stream>>>(ctx, Wo, bo, (void*)outp);
}
